// GAT_GN_71004399337806
// MI455X (gfx1250) — matrix operand order repaired and checked by reading, not compiled, not run on hardware
//
#include <hip/hip_runtime.h>
#include <hip/hip_bf16.h>
#include <cmath>
#include <cstdint>

// ---------------- problem constants (match reference) ----------------
#define N_NODES 20000
#define N_EDGES 320000
#define ETOT    (N_EDGES + N_NODES)   // edges + self loops
#define NF      128
#define HID     128
#define NHEADS  4
#define OUTD    100

// GEMM tiling
#define BM 128
#define BN 64
#define BK 32

// ---------------- CDNA5 vector types ----------------
typedef __attribute__((ext_vector_type(16))) __bf16 v16bf;
typedef __attribute__((ext_vector_type(8)))  __bf16 v8bf;
typedef __attribute__((ext_vector_type(8)))  float  v8f;

static __device__ __forceinline__ uint16_t f32_to_bf16_rne(float f) {
    uint32_t u = __float_as_uint(f);
    uint32_t r = 0x7FFFu + ((u >> 16) & 1u);
    return (uint16_t)((u + r) >> 16);
}

// Async global->LDS 16B copy (per-lane). VDST VGPR carries the LDS byte
// offset (low 32 bits of the flat shared pointer ARE the LDS address on
// gfx1250 per the aperture mapping), VADDR is the 64-bit global address.
// Tracked by ASYNCcnt.
static __device__ __forceinline__ void async_copy_b128(const void* gp, void* lp) {
    uint32_t lds_off = (uint32_t)(uintptr_t)lp;
    asm volatile("global_load_async_to_lds_b128 %0, %1, off"
                 :: "v"(lds_off), "v"(gp) : "memory");
}
static __device__ __forceinline__ void wait_asynccnt0() {
    asm volatile("s_wait_asynccnt 0x0" ::: "memory");
}

// ---------------- elementwise kernels ----------------
__global__ void cast_f32_to_bf16(const float* __restrict__ in,
                                 uint16_t* __restrict__ out, int n) {
    int i = blockIdx.x * blockDim.x + threadIdx.x;
    if (i < n) out[i] = f32_to_bf16_rne(in[i]);
}

__global__ void fill_f32(float* __restrict__ p, float v, int n) {
    int i = blockIdx.x * blockDim.x + threadIdx.x;
    if (i < n) p[i] = v;
}

// ---------------- WMMA GEMM with async-LDS staging -------------------
// C[M,N] = A[M,K](bf16) * W[N,K](bf16)^T, f32 out. 256 thr = 8 waves.
// Per block: BM x BN output, K consumed in BK steps.
//  - stage: As[128][32], Bs[64][32] via global_load_async_to_lds_b128
//  - wave w computes rows [w*16, w*16+16) x all BN columns (4 WMMA tiles)
// Fragment layouts used here (16-bit A 16x32 / B 32x16, contiguous k; A and B match):
//  A lane(g,r): A[row, kb+16g..+15] (one contiguous 32B chunk)
//  B lane(g,r): W[col, kb+16g..+15] (one contiguous 32B chunk)
__global__ __launch_bounds__(256)
void wmma_gemm_bf16_lds(const uint16_t* __restrict__ A,
                        const uint16_t* __restrict__ W,
                        float* __restrict__ C,
                        int M, int N, int K) {
    __shared__ __align__(64) uint16_t As[BM * BK];   // 8 KB
    __shared__ __align__(64) uint16_t Bs[BN * BK];   // 4 KB

    const int tilesN = (N + BN - 1) / BN;
    const int bm0 = (blockIdx.x / tilesN) * BM;
    const int bn0 = (blockIdx.x % tilesN) * BN;
    const int tid  = threadIdx.x;
    const int wv   = tid >> 5;
    const int lane = tid & 31;
    const int g = lane >> 4, r = lane & 15;

    v8f acc[BN / 16] = {};

    for (int kb = 0; kb < K; kb += BK) {
        // ---- async stage: A = 512 x 16B chunks, B = 256 x 16B chunks ----
        #pragma unroll
        for (int it = 0; it < 2; ++it) {
            int c   = tid + it * 256;          // chunk id: row = c/4, 16B sub = c%4
            int row = c >> 2, cc = c & 3;
            int gr  = bm0 + row; if (gr >= M) gr = M - 1;   // clamp (store-guarded)
            async_copy_b128(A + (size_t)gr * K + kb + cc * 8, &As[c * 8]);
        }
        {
            int c   = tid;
            int row = c >> 2, cc = c & 3;
            int gr  = bn0 + row; if (gr >= N) gr = N - 1;   // clamp (store-guarded)
            async_copy_b128(W + (size_t)gr * K + kb + cc * 8, &Bs[c * 8]);
        }
        __builtin_prefetch(A + (size_t)(bm0 + (tid >> 1)) * K + kb + BK, 0, 0);
        wait_asynccnt0();
        __syncthreads();

        // ---- compute: 1 A fragment, 4 B fragments, 4 WMMAs per wave ----
        {
            const uint16_t* arow = &As[(wv * 16 + r) * BK];
            v16bf a  = *(const v16bf*)(arow + 16 * g);
            #pragma unroll
            for (int n0 = 0; n0 < BN / 16; ++n0) {
                v16bf b = *(const v16bf*)(&Bs[(n0 * 16 + r) * BK + 16 * g]);
                acc[n0] = __builtin_amdgcn_wmma_f32_16x16x32_bf16(
                              false, a, false, b, (short)0, acc[n0], false, false);
            }
        }
        __syncthreads();
    }

    // ---- store: lane(g,r), VGPR v -> D[8g+v][r] of each 16x16 tile ----
    #pragma unroll
    for (int n0 = 0; n0 < BN / 16; ++n0) {
        int ncol = bn0 + n0 * 16 + r;
        if (ncol < N) {
            #pragma unroll
            for (int v = 0; v < 8; ++v) {
                int mrow = bm0 + wv * 16 + 8 * g + v;
                if (mrow < M) C[(size_t)mrow * N + ncol] = acc[n0][v];
            }
        }
    }
}

// ---------------- per-(node,head) attention logits ----------------
__global__ void attn_scores(const float* __restrict__ h,
                            const float* __restrict__ att,
                            float* __restrict__ ai, float* __restrict__ aj,
                            int n_nodes, int heads, int C) {
    int wave = (blockIdx.x * blockDim.x + threadIdx.x) >> 5;
    int lane = threadIdx.x & 31;
    if (wave >= n_nodes * heads) return;
    int n = wave / heads, hh = wave % heads;
    const float* __restrict__ hv = h + (size_t)n * heads * C + (size_t)hh * C;
    const float* __restrict__ at = att + (size_t)hh * 2 * C;
    float si = 0.f, sj = 0.f;
    for (int c = lane; c < C; c += 32) {
        float v = hv[c];
        si += v * at[c];
        sj += v * at[C + c];
    }
    #pragma unroll
    for (int off = 16; off > 0; off >>= 1) {
        si += __shfl_down(si, off, 32);
        sj += __shfl_down(sj, off, 32);
    }
    if (lane == 0) { ai[wave] = si; aj[wave] = sj; }
}

// ---------------- edge helpers ----------------
static __device__ __forceinline__ void edge_sd(const int* __restrict__ ei,
                                               int e, int& s, int& d) {
    if (e < N_EDGES) { s = ei[e]; d = ei[N_EDGES + e]; }
    else             { s = d = e - N_EDGES; }          // self loop
}

static __device__ __forceinline__ void atomic_max_f32(float* addr, float v) {
    if (v >= 0.f) atomicMax((int*)addr, __float_as_int(v));
    else          atomicMin((unsigned int*)addr, __float_as_uint(v));
}

// pass 1: segment max of alpha over dst
__global__ void edge_max(const int* __restrict__ ei,
                         const float* __restrict__ ai,
                         const float* __restrict__ aj,
                         float* __restrict__ m, int heads) {
    int t = blockIdx.x * blockDim.x + threadIdx.x;
    if (t >= ETOT * heads) return;
    int e = t / heads, hh = t % heads;
    int s, d; edge_sd(ei, e, s, d);
    float alpha = ai[d * heads + hh] + aj[s * heads + hh];
    atomic_max_f32(&m[d * heads + hh], alpha);
}

// pass 2: e = exp(alpha - m[dst]); segment sum; stash e in coef
__global__ void edge_expsum(const int* __restrict__ ei,
                            const float* __restrict__ ai,
                            const float* __restrict__ aj,
                            const float* __restrict__ m,
                            float* __restrict__ ssum,
                            float* __restrict__ coef, int heads) {
    int t = blockIdx.x * blockDim.x + threadIdx.x;
    if (t >= ETOT * heads) return;
    int e = t / heads, hh = t % heads;
    int s, d; edge_sd(ei, e, s, d);
    float alpha = ai[d * heads + hh] + aj[s * heads + hh];
    float ev = __expf(alpha - m[d * heads + hh]);
    coef[t] = ev;
    atomicAdd(&ssum[d * heads + hh], ev);
}

// pass 3: agg[dst] += h[src] * (e / s[dst]) ; one thread per (edge,head,4ch)
__global__ void edge_scatter(const int* __restrict__ ei,
                             const float* __restrict__ coef,
                             const float* __restrict__ ssum,
                             const float* __restrict__ h,
                             float* __restrict__ agg, int heads, int C) {
    int chunks = C >> 2;
    long long t = (long long)blockIdx.x * blockDim.x + threadIdx.x;
    if (t >= (long long)ETOT * heads * chunks) return;
    int c4 = (int)(t % chunks);
    int eh = (int)(t / chunks);
    int hh = eh % heads, e = eh / heads;
    int s, d; edge_sd(ei, e, s, d);
    float a = coef[eh] / ssum[d * heads + hh];
    const float4 msg = *(const float4*)(h + (size_t)s * heads * C + (size_t)hh * C + 4 * c4);
    float* outp = agg + (size_t)d * heads * C + (size_t)hh * C + 4 * c4;
    atomicAdd(outp + 0, msg.x * a);
    atomicAdd(outp + 1, msg.y * a);
    atomicAdd(outp + 2, msg.z * a);
    atomicAdd(outp + 3, msg.w * a);
}

// ---------------- host driver ----------------
extern "C" void kernel_launch(void* const* d_in, const int* in_sizes, int n_in,
                              void* d_out, int out_size, void* d_ws, size_t ws_size,
                              hipStream_t stream) {
    (void)in_sizes; (void)n_in; (void)out_size; (void)ws_size;
    const float* x    = (const float*)d_in[0];
    const float* W1   = (const float*)d_in[1];
    const float* att1 = (const float*)d_in[2];
    const float* W2   = (const float*)d_in[3];
    const float* att2 = (const float*)d_in[4];
    const float* W3   = (const float*)d_in[5];
    const float* att3 = (const float*)d_in[6];
    const int*   ei   = (const int*)d_in[7];
    float* out = (float*)d_out;

    char* p = (char*)d_ws;
    auto take = [&](size_t bytes) -> char* {
        char* q = p; p += (bytes + 255) & ~(size_t)255; return q;
    };
    uint16_t* hbf  = (uint16_t*)take((size_t)N_NODES * 512 * 2);
    uint16_t* wbf  = (uint16_t*)take((size_t)512 * 512 * 2);
    float*    hlin = (float*)take((size_t)N_NODES * 512 * 4);
    float*    agg  = (float*)take((size_t)N_NODES * 512 * 4);
    float*    ai   = (float*)take((size_t)N_NODES * NHEADS * 4);
    float*    aj   = (float*)take((size_t)N_NODES * NHEADS * 4);
    float*    mbuf = (float*)take((size_t)N_NODES * NHEADS * 4);
    float*    sbuf = (float*)take((size_t)N_NODES * NHEADS * 4);
    float*    coef = (float*)take((size_t)ETOT * NHEADS * 4);

    auto cdiv = [](long long a, long long b) { return (int)((a + b - 1) / b); };

    auto run_layer = [&](const float* hin, const float* Wf, const float* attf,
                         int inK, int heads, int C, float* aggout) {
        const int M = N_NODES, Nout = heads * C;
        // 1) stage bf16 operands
        cast_f32_to_bf16<<<cdiv(M * inK, 256), 256, 0, stream>>>(hin, hbf, M * inK);
        cast_f32_to_bf16<<<cdiv(Nout * inK, 256), 256, 0, stream>>>(Wf, wbf, Nout * inK);
        // 2) WMMA projection -> hlin (async-LDS staged, BMxBN per block)
        {
            int blocks = cdiv(M, BM) * cdiv(Nout, BN);
            wmma_gemm_bf16_lds<<<blocks, 256, 0, stream>>>(hbf, wbf, hlin, M, Nout, inK);
        }
        // 3) per-(node,head) attention logits
        attn_scores<<<cdiv((long long)M * heads, 8), 256, 0, stream>>>(
            hlin, attf, ai, aj, M, heads, C);
        // 4) init softmax state + output accumulator
        fill_f32<<<cdiv(M * heads, 256), 256, 0, stream>>>(mbuf, -INFINITY, M * heads);
        fill_f32<<<cdiv(M * heads, 256), 256, 0, stream>>>(sbuf, 0.f, M * heads);
        fill_f32<<<cdiv((long long)M * Nout, 256), 256, 0, stream>>>(aggout, 0.f, M * Nout);
        // 5) segment softmax + scatter-aggregate
        {
            int t12 = ETOT * heads;
            edge_max   <<<cdiv(t12, 256), 256, 0, stream>>>(ei, ai, aj, mbuf, heads);
            edge_expsum<<<cdiv(t12, 256), 256, 0, stream>>>(ei, ai, aj, mbuf, sbuf, coef, heads);
            long long t3 = (long long)ETOT * heads * (C / 4);
            edge_scatter<<<cdiv(t3, 256), 256, 0, stream>>>(ei, coef, sbuf, hlin, aggout, heads, C);
        }
    };

    // layer 1: [N,128] -> heads=4, C=128
    run_layer(x,   W1, att1, NF,           NHEADS, HID,  agg);
    // layer 2: [N,512] -> heads=4, C=128 (agg reuse safe under stream ordering)
    run_layer(agg, W2, att2, NHEADS * HID, NHEADS, HID,  agg);
    // layer 3: [N,512] -> heads=1, C=100 -> d_out
    run_layer(agg, W3, att3, NHEADS * HID, 1,      OUTD, out);
}
